// GAT_85581518340590
// MI455X (gfx1250) — matrix operand order repaired and checked by reading, not compiled, not run on hardware
//
#include <hip/hip_runtime.h>
#include <hip/hip_bf16.h>
#include <math.h>

// ---------------------------------------------------------------------------
// Model constants (from reference)
// ---------------------------------------------------------------------------
#define NTOK   4096          // B*S
#define BATCH  2
#define SQ     2048
#define DMODEL 1024
#define NHEAD  16
#define HDIM   64
#define FFDIM  4096
#define NLAYER 4
#define VOCAB  18691
#define SOFTCAP 30.0f

typedef __bf16 bf16_t;
typedef bf16_t v16bf __attribute__((ext_vector_type(16)));
typedef bf16_t v8bf  __attribute__((ext_vector_type(8)));
typedef float  v8f   __attribute__((ext_vector_type(8)));

static __device__ inline v16bf mk16(v8bf lo, v8bf hi) {
  v16bf r;
#pragma unroll
  for (int i = 0; i < 8; ++i) { r[i] = lo[i]; r[i + 8] = hi[i]; }
  return r;
}

static __device__ inline v8f wmma_bf16(v16bf a, v16bf b, v8f c) {
  // D = A(16x32) * B(32x16) + C, f32 accumulate
  return __builtin_amdgcn_wmma_f32_16x16x32_bf16(false, a, false, b,
                                                 (short)0, c, false, false);
}

static __device__ inline v8f vzero8() {
  v8f z = {0.f, 0.f, 0.f, 0.f, 0.f, 0.f, 0.f, 0.f};
  return z;
}

// CDNA5 async global->LDS copy (ASYNCcnt-tracked), 16B per lane.
static __device__ inline void async_copy_b128(unsigned lds_byte_off,
                                              const void* gptr) {
  asm volatile("global_load_async_to_lds_b128 %0, %1, off"
               :: "v"(lds_byte_off), "v"(gptr) : "memory");
}
static __device__ inline void wait_asynccnt0() {
  asm volatile("s_wait_asynccnt 0" ::: "memory");
}
static __device__ inline void wait_dscnt0() {
  asm volatile("s_wait_dscnt 0" ::: "memory");
}

static __device__ inline float wave_sum32(float v) {
  v += __shfl_xor(v, 1, 32);  v += __shfl_xor(v, 2, 32);
  v += __shfl_xor(v, 4, 32);  v += __shfl_xor(v, 8, 32);
  v += __shfl_xor(v, 16, 32);
  return v;
}
static __device__ inline float wave_sum16(float v) {
  v += __shfl_xor(v, 1, 32);  v += __shfl_xor(v, 2, 32);
  v += __shfl_xor(v, 4, 32);  v += __shfl_xor(v, 8, 32);
  return v;
}
static __device__ inline float wave_max16(float v) {
  v = fmaxf(v, __shfl_xor(v, 1, 32));  v = fmaxf(v, __shfl_xor(v, 2, 32));
  v = fmaxf(v, __shfl_xor(v, 4, 32));  v = fmaxf(v, __shfl_xor(v, 8, 32));
  return v;
}

// ---------------------------------------------------------------------------
// Embedding gather + rms_norm + level computation
// ---------------------------------------------------------------------------
__global__ __launch_bounds__(256)
void embed_kernel(const int* __restrict__ idx, const float* __restrict__ wte,
                  float* __restrict__ x, float* __restrict__ x0,
                  int* __restrict__ levels) {
  int tok = blockIdx.x;
  int t = threadIdx.x;
  int id = idx[tok];
  const float* row = wte + (size_t)id * DMODEL;
  float v[4];
  float ss = 0.f;
#pragma unroll
  for (int j = 0; j < 4; ++j) { v[j] = row[t + j * 256]; ss += v[j] * v[j]; }
  __shared__ float sred[8];
  float ws = wave_sum32(ss);
  if ((t & 31) == 0) sred[t >> 5] = ws;
  __syncthreads();
  float tot = 0.f;
#pragma unroll
  for (int i = 0; i < 8; ++i) tot += sred[i];
  float rs = rsqrtf(tot / (float)DMODEL + 1e-6f);
#pragma unroll
  for (int j = 0; j < 4; ++j) {
    float o = v[j] * rs;
    x[(size_t)tok * DMODEL + t + j * 256] = o;
    x0[(size_t)tok * DMODEL + t + j * 256] = o;
  }
  if (t == 0)
    levels[tok] = (id >= 16385) + (id >= 18434) + (id >= 18691);
}

// ---------------------------------------------------------------------------
// xr = l0*x + l1*x0 ; h = rms_norm(xr)
// ---------------------------------------------------------------------------
__global__ __launch_bounds__(256)
void resid_rms_kernel(const float* __restrict__ x, const float* __restrict__ x0,
                      const float* __restrict__ lambdas, int layer,
                      float* __restrict__ xr, float* __restrict__ h) {
  int tok = blockIdx.x;
  int t = threadIdx.x;
  float l0 = lambdas[layer * 2 + 0];
  float l1 = lambdas[layer * 2 + 1];
  float v[4];
  float ss = 0.f;
#pragma unroll
  for (int j = 0; j < 4; ++j) {
    size_t e = (size_t)tok * DMODEL + t + j * 256;
    float a = l0 * x[e] + l1 * x0[e];
    v[j] = a;
    xr[e] = a;
    ss += a * a;
  }
  __shared__ float sred[8];
  float ws = wave_sum32(ss);
  if ((t & 31) == 0) sred[t >> 5] = ws;
  __syncthreads();
  float tot = 0.f;
#pragma unroll
  for (int i = 0; i < 8; ++i) tot += sred[i];
  float rs = rsqrtf(tot / (float)DMODEL + 1e-6f);
#pragma unroll
  for (int j = 0; j < 4; ++j)
    h[(size_t)tok * DMODEL + t + j * 256] = v[j] * rs;
}

// plain rms_norm: h = rms_norm(x)
__global__ __launch_bounds__(256)
void rms_kernel(const float* __restrict__ x, float* __restrict__ h) {
  int tok = blockIdx.x;
  int t = threadIdx.x;
  float v[4];
  float ss = 0.f;
#pragma unroll
  for (int j = 0; j < 4; ++j) {
    v[j] = x[(size_t)tok * DMODEL + t + j * 256];
    ss += v[j] * v[j];
  }
  __shared__ float sred[8];
  float ws = wave_sum32(ss);
  if ((t & 31) == 0) sred[t >> 5] = ws;
  __syncthreads();
  float tot = 0.f;
#pragma unroll
  for (int i = 0; i < 8; ++i) tot += sred[i];
  float rs = rsqrtf(tot / (float)DMODEL + 1e-6f);
#pragma unroll
  for (int j = 0; j < 4; ++j)
    h[(size_t)tok * DMODEL + t + j * 256] = v[j] * rs;
}

// ---------------------------------------------------------------------------
// Generic WMMA GEMM:  C[M,N] = A[M,K] * W[N,K]^T   (+ epilogue)
// EPI 0: C = acc          (f32)
// EPI 1: C = Res + acc    (f32 residual add)
// EPI 2: C = relu(acc)^2  (f32)
// Block tile 128x64 (BK=32), 8 waves in 4x2, each wave a 32x32 accumulator
// (2 A-frags x 2 B-frags -> 4 WMMAs per K-step).
// f32 global -> bf16 LDS convert while staging.
// ---------------------------------------------------------------------------
#define LDSTR 48  // padded LDS row stride (bf16 elems); 96B keeps 32B alignment

template <int EPI>
__global__ __launch_bounds__(256)
void gemm_kernel(const float* __restrict__ A, const float* __restrict__ W,
                 const float* __restrict__ Res, float* __restrict__ C,
                 int M, int N, int K) {
  __shared__ bf16_t As[128][LDSTR];
  __shared__ bf16_t Bs[64][LDSTR];
  int bn = blockIdx.x, bm = blockIdx.y;
  int tid = threadIdx.x;
  int lane = tid & 31, wid = tid >> 5;
  int wr = wid >> 1, wc = wid & 1;       // wave tile: rows wr*32, cols wc*32
  int n = lane & 15, g = lane >> 4;
  int arow = tid >> 1, acol = (tid & 1) * 16;  // A staging: 16 f32 / thread
  int brow = tid >> 2, bcol = (tid & 3) * 8;   // B staging: 8 f32 / thread

  v8f cc[2][2] = {{vzero8(), vzero8()}, {vzero8(), vzero8()}};
  const float* aBase = A + (size_t)(bm * 128 + arow) * K + acol;
  const float* bBase = W + (size_t)(bn * 64 + brow) * K + bcol;

  for (int k0 = 0; k0 < K; k0 += 32) {
#pragma unroll
    for (int j = 0; j < 16; ++j) As[arow][acol + j] = (bf16_t)aBase[k0 + j];
#pragma unroll
    for (int j = 0; j < 8; ++j)  Bs[brow][bcol + j] = (bf16_t)bBase[k0 + j];
    if (k0 + 32 < K) {                   // global_prefetch_b8 for next tile
      __builtin_prefetch(aBase + k0 + 32, 0, 1);
      __builtin_prefetch(bBase + k0 + 32, 0, 1);
    }
    __syncthreads();

    // A fragments (two 16-row subtiles): e<8 -> k=e+8g ; e>=8 -> k=e+8+8g
    v16bf a[2], b[2];
#pragma unroll
    for (int rt = 0; rt < 2; ++rt) {
      int row = wr * 32 + rt * 16 + n;
      v8bf lo = *(const v8bf*)&As[row][8 * g];
      v8bf hi = *(const v8bf*)&As[row][8 * g + 16];
      a[rt] = mk16(lo, hi);
    }
    // B fragments (two 16-col subtiles): e<8 -> k=e+8g ; e>=8 -> k=e+8+8g
#pragma unroll
    for (int grp = 0; grp < 2; ++grp) {
      int col = wc * 32 + grp * 16 + n;
      v8bf lo = *(const v8bf*)&Bs[col][8 * g];
      v8bf hi = *(const v8bf*)&Bs[col][8 * g + 16];
      b[grp] = mk16(lo, hi);
    }
#pragma unroll
    for (int rt = 0; rt < 2; ++rt)
#pragma unroll
      for (int grp = 0; grp < 2; ++grp)
        cc[rt][grp] = wmma_bf16(a[rt], b[grp], cc[rt][grp]);
    __syncthreads();
  }

#pragma unroll
  for (int rt = 0; rt < 2; ++rt) {
#pragma unroll
    for (int r = 0; r < 8; ++r) {
      int row = bm * 128 + wr * 32 + rt * 16 + r + 8 * g;
#pragma unroll
      for (int grp = 0; grp < 2; ++grp) {
        int col = bn * 64 + wc * 32 + grp * 16 + n;
        float val = cc[rt][grp][r];
        size_t o = (size_t)row * N + col;
        if (EPI == 0) {
          C[o] = val;
        } else if (EPI == 1) {
          C[o] = Res[o] + val;
        } else {
          float rv = fmaxf(val, 0.f);
          C[o] = rv * rv;
        }
      }
    }
  }
}

// ---------------------------------------------------------------------------
// Per-(token,head): rms_norm+rotary for q,k -> bf16 [B,H,S,64];
// v blend with v1 -> transposed bf16 [B,H,64,S]
// ---------------------------------------------------------------------------
__global__ __launch_bounds__(32)
void qk_prep_kernel(const float* __restrict__ qf, const float* __restrict__ kf,
                    const float* __restrict__ vf, float* __restrict__ v1f,
                    const float* __restrict__ lambp, int layer,
                    bf16_t* __restrict__ qbf, bf16_t* __restrict__ kbf,
                    bf16_t* __restrict__ vtbf) {
  int tok = blockIdx.x, h = blockIdx.y;
  int b = tok / SQ, s = tok % SQ;
  int t = threadIdx.x;  // 0..31 == rotary pair index
  size_t base = (size_t)tok * DMODEL + h * HDIM;
  size_t obase = (((size_t)b * NHEAD + h) * SQ + s) * HDIM;

  float inv = powf(10000.f, -(float)t / 32.f);
  float ang = (float)s * inv;
  float cs = cosf(ang), sn = sinf(ang);

#pragma unroll
  for (int which = 0; which < 2; ++which) {
    const float* src = which ? kf : qf;
    bf16_t* dst = which ? kbf : qbf;
    float x1 = src[base + t];
    float x2 = src[base + t + 32];
    float ss = wave_sum32(x1 * x1 + x2 * x2);
    float rs = rsqrtf(ss / (float)HDIM + 1e-6f);
    x1 *= rs; x2 *= rs;
    dst[obase + t]      = (bf16_t)(x1 * cs + x2 * sn);
    dst[obase + t + 32] = (bf16_t)(-x1 * sn + x2 * cs);
  }

  float v0 = vf[base + t];
  float v0b = vf[base + t + 32];
  float w1v, w2v;
  if (layer == 0) {
    v1f[base + t] = v0; v1f[base + t + 32] = v0b;
    w1v = v0; w2v = v0b;
  } else {
    w1v = v1f[base + t]; w2v = v1f[base + t + 32];
  }
  float lam = lambp[layer];
  float b1 = (1.f - lam) * v0 + lam * w1v;
  float b2 = (1.f - lam) * v0b + lam * w2v;
  size_t vb = ((size_t)b * NHEAD + h) * HDIM;
  vtbf[(vb + t) * SQ + s]      = (bf16_t)b1;
  vtbf[(vb + t + 32) * SQ + s] = (bf16_t)b2;
}

// ---------------------------------------------------------------------------
// Flash attention, one wave per (batch, head, 16-query block).
// K/V tiles staged via CDNA5 async global->LDS copies (ASYNCcnt),
// scores = Q K^T / 8 with mask (causal & (level_k>0 | q-k<=256)),
// online softmax, P staged through LDS as bf16, P*V via WMMA.
// ---------------------------------------------------------------------------
__global__ __launch_bounds__(32)
void attn_kernel(const bf16_t* __restrict__ qbf, const bf16_t* __restrict__ kbf,
                 const bf16_t* __restrict__ vtbf, const int* __restrict__ levels,
                 float* __restrict__ y) {
  int qb = blockIdx.x, h = blockIdx.y, b = blockIdx.z;
  int lane = threadIdx.x;
  int n = lane & 15, g = lane >> 4;
  __shared__ bf16_t Ks[32][64];   // K tile: 32 keys x 64 dims (4KB)
  __shared__ bf16_t Vs[64][32];   // V tile: 64 dims x 32 keys (4KB)
  __shared__ bf16_t Ps[16][32];   // probabilities (1KB)

  size_t bh = (size_t)b * NHEAD + h;
  const bf16_t* qB = qbf + bh * SQ * HDIM;
  const bf16_t* kB = kbf + bh * SQ * HDIM;
  const bf16_t* vB = vtbf + bh * HDIM * SQ;

  unsigned ksBase = (unsigned)(uintptr_t)&Ks[0][0];  // LDS byte offsets
  unsigned vsBase = (unsigned)(uintptr_t)&Vs[0][0];

  // Q A-fragments (rows = qb*16 + n), two K-steps of 32 over head dim
  v16bf aq[2];
#pragma unroll
  for (int s = 0; s < 2; ++s) {
    const bf16_t* qp = qB + (size_t)(qb * 16 + n) * HDIM + 32 * s + 8 * g;
    v8bf lo = *(const v8bf*)qp;
    v8bf hi = *(const v8bf*)(qp + 16);
    aq[s] = mk16(lo, hi);
  }

  v8f acc[4] = {vzero8(), vzero8(), vzero8(), vzero8()};
  float mrow[8], lrow[8];
#pragma unroll
  for (int r = 0; r < 8; ++r) { mrow[r] = -INFINITY; lrow[r] = 0.f; }

  int kbmax = (qb * 16 + 15) >> 5;
  for (int kb = 0; kb <= kbmax; ++kb) {
    // ---- async-stage K tile (contiguous 4KB) and V tile (64 rows x 64B) ----
    const char* ksrc = (const char*)(kB + (size_t)kb * 32 * HDIM);
#pragma unroll
    for (int c = 0; c < 8; ++c) {
      unsigned off = (unsigned)((lane + c * 32) * 16);
      async_copy_b128(ksBase + off, ksrc + off);
    }
#pragma unroll
    for (int c = 0; c < 8; ++c) {
      int chunk = lane + c * 32;           // 0..255
      int r = chunk >> 2, co = (chunk & 3) * 16;
      async_copy_b128(vsBase + (unsigned)(r * 64 + co),
                      (const char*)vB + (size_t)r * (SQ * 2) +
                          (size_t)kb * 64 + co);
    }
    wait_asynccnt0();

    // ---- scores: 16 x 32 (two 16x16 WMMA groups) ----
    v8f c[2] = {vzero8(), vzero8()};
#pragma unroll
    for (int grp = 0; grp < 2; ++grp) {
#pragma unroll
      for (int s = 0; s < 2; ++s) {
        v8bf lo = *(const v8bf*)&Ks[grp * 16 + n][32 * s + 8 * g];
        v8bf hi = *(const v8bf*)&Ks[grp * 16 + n][32 * s + 8 * g + 16];
        c[grp] = wmma_bf16(aq[s], mk16(lo, hi), c[grp]);
      }
    }
    int lev0 = levels[b * SQ + kb * 32 + n];
    int lev1 = levels[b * SQ + kb * 32 + 16 + n];

#pragma unroll
    for (int r = 0; r < 8; ++r) {
      int q = qb * 16 + r + 8 * g;
      float sv[2];
      float rowmax = -INFINITY;
#pragma unroll
      for (int grp = 0; grp < 2; ++grp) {
        int key = kb * 32 + grp * 16 + n;
        float sc = c[grp][r] * 0.125f;
        int lev = grp ? lev1 : lev0;
        bool ok = (q >= key) && ((lev > 0) || (q - key <= 256));
        sc = ok ? sc : -INFINITY;
        sv[grp] = sc;
        rowmax = fmaxf(rowmax, sc);
      }
      rowmax = wave_max16(rowmax);
      float mn = fmaxf(mrow[r], rowmax);
      float alpha = (mrow[r] <= -1e37f || mn <= -1e37f)
                        ? 0.f : expf(mrow[r] - mn);
      float p0 = (sv[0] <= -1e37f || mn <= -1e37f) ? 0.f : expf(sv[0] - mn);
      float p1 = (sv[1] <= -1e37f || mn <= -1e37f) ? 0.f : expf(sv[1] - mn);
      float psum = wave_sum16(p0 + p1);
      lrow[r] = lrow[r] * alpha + psum;
      mrow[r] = mn;
#pragma unroll
      for (int d = 0; d < 4; ++d) acc[d][r] *= alpha;
      Ps[r + 8 * g][n]      = (bf16_t)p0;
      Ps[r + 8 * g][16 + n] = (bf16_t)p1;
    }
    // make P visible for LDS fragment reads (same wave, split DS counter)
    wait_dscnt0();

    // ---- P(16x32) * V(32x64) ----
    v8bf plo = *(const v8bf*)&Ps[n][8 * g];
    v8bf phi = *(const v8bf*)&Ps[n][8 * g + 16];
    v16bf ap = mk16(plo, phi);
#pragma unroll
    for (int d = 0; d < 4; ++d) {
      v8bf lo = *(const v8bf*)&Vs[d * 16 + n][8 * g];
      v8bf hi = *(const v8bf*)&Vs[d * 16 + n][8 * g + 16];
      acc[d] = wmma_bf16(ap, mk16(lo, hi), acc[d]);
    }
  }

#pragma unroll
  for (int r = 0; r < 8; ++r) {
    int q = qb * 16 + r + 8 * g;
    float linv = 1.f / lrow[r];
#pragma unroll
    for (int d = 0; d < 4; ++d) {
      y[((size_t)(b * SQ + q)) * DMODEL + h * HDIM + d * 16 + n] =
          acc[d][r] * linv;
    }
  }
}

// ---------------------------------------------------------------------------
// lm_head WMMA GEMM (128x64 tile) with fused softcap + exp-sum
// (max fixed at SOFTCAP) + target-logit capture.
// sumexp[token] += sum_v exp(softcap(l) - 30)
// ---------------------------------------------------------------------------
__global__ __launch_bounds__(256)
void lm_kernel(const float* __restrict__ A, const float* __restrict__ W,
               const int* __restrict__ target, float* __restrict__ sumexp,
               float* __restrict__ tgtlog, int M, int N, int K) {
  __shared__ bf16_t As[128][LDSTR];
  __shared__ bf16_t Bs[64][LDSTR];
  int bn = blockIdx.x, bm = blockIdx.y;
  int tid = threadIdx.x;
  int lane = tid & 31, wid = tid >> 5;
  int wr = wid >> 1, wc = wid & 1;
  int n = lane & 15, g = lane >> 4;
  int arow = tid >> 1, acol = (tid & 1) * 16;
  int brow = tid >> 2, bcol = (tid & 3) * 8;

  v8f cc[2][2] = {{vzero8(), vzero8()}, {vzero8(), vzero8()}};
  int br = bn * 64 + brow;
  if (br >= N) br = N - 1;  // clamp (epilogue drops cols >= N)
  const float* aBase = A + (size_t)(bm * 128 + arow) * K + acol;
  const float* bBase = W + (size_t)br * K + bcol;

  for (int k0 = 0; k0 < K; k0 += 32) {
#pragma unroll
    for (int j = 0; j < 16; ++j) As[arow][acol + j] = (bf16_t)aBase[k0 + j];
#pragma unroll
    for (int j = 0; j < 8; ++j)  Bs[brow][bcol + j] = (bf16_t)bBase[k0 + j];
    if (k0 + 32 < K) {
      __builtin_prefetch(aBase + k0 + 32, 0, 1);
      __builtin_prefetch(bBase + k0 + 32, 0, 1);
    }
    __syncthreads();
    v16bf a[2], b[2];
#pragma unroll
    for (int rt = 0; rt < 2; ++rt) {
      int row = wr * 32 + rt * 16 + n;
      v8bf lo = *(const v8bf*)&As[row][8 * g];
      v8bf hi = *(const v8bf*)&As[row][8 * g + 16];
      a[rt] = mk16(lo, hi);
    }
#pragma unroll
    for (int grp = 0; grp < 2; ++grp) {
      int col = wc * 32 + grp * 16 + n;
      v8bf lo = *(const v8bf*)&Bs[col][8 * g];
      v8bf hi = *(const v8bf*)&Bs[col][8 * g + 16];
      b[grp] = mk16(lo, hi);
    }
#pragma unroll
    for (int rt = 0; rt < 2; ++rt)
#pragma unroll
      for (int grp = 0; grp < 2; ++grp)
        cc[rt][grp] = wmma_bf16(a[rt], b[grp], cc[rt][grp]);
    __syncthreads();
  }

#pragma unroll
  for (int rt = 0; rt < 2; ++rt) {
#pragma unroll
    for (int r = 0; r < 8; ++r) {
      int row = bm * 128 + wr * 32 + rt * 16 + r + 8 * g;
      int tg = target[row];
      float part = 0.f;
#pragma unroll
      for (int grp = 0; grp < 2; ++grp) {
        int col = bn * 64 + wc * 32 + grp * 16 + n;
        float val = cc[rt][grp][r];
        float sc = SOFTCAP * tanhf(val / SOFTCAP);
        if (col < N) {
          part += expf(sc - SOFTCAP);
          if (col == tg) tgtlog[row] = sc;
        }
      }
      part = wave_sum16(part);
      if (n == 0) atomicAdd(&sumexp[row], part);
    }
  }
}

__global__ void loss_kernel(const int* __restrict__ target,
                            const float* __restrict__ sumexp,
                            const float* __restrict__ tgtlog,
                            float* __restrict__ out, int ntok) {
  int i = blockIdx.x * blockDim.x + threadIdx.x;
  if (i >= ntok) return;
  int t = target[i];
  if (t == 16384 || t == 18433 || t == 18690) {
    out[i] = 0.f;
  } else {
    float lse = SOFTCAP + logf(sumexp[i]);
    out[i] = lse - tgtlog[i];
  }
}

// ---------------------------------------------------------------------------
// Host-side orchestration
// ---------------------------------------------------------------------------
extern "C" void kernel_launch(void* const* d_in, const int* in_sizes, int n_in,
                              void* d_out, int out_size, void* d_ws,
                              size_t ws_size, hipStream_t stream) {
  (void)in_sizes; (void)n_in; (void)out_size; (void)ws_size;
  const int*   idx     = (const int*)d_in[0];
  const int*   target  = (const int*)d_in[1];
  const float* wte     = (const float*)d_in[2];
  const float* lambdas = (const float*)d_in[3];
  const float* lamb    = (const float*)d_in[4];
  const float* wq      = (const float*)d_in[5];
  const float* wk      = (const float*)d_in[6];
  const float* wv      = (const float*)d_in[7];
  const float* wo      = (const float*)d_in[8];
  const float* w1      = (const float*)d_in[9];
  const float* w2      = (const float*)d_in[10];
  const float* lm_head = (const float*)d_in[11];

  // workspace carve-out
  char* w = (char*)d_ws;
  size_t off = 0;
  auto carve = [&](size_t bytes) -> char* {
    char* p = w + off;
    off = (off + bytes + 255) & ~(size_t)255;
    return p;
  };
  const size_t ND4 = (size_t)NTOK * DMODEL * 4;
  const size_t ND2 = (size_t)NTOK * DMODEL * 2;
  float* x    = (float*)carve(ND4);
  float* x0   = (float*)carve(ND4);
  float* xr   = (float*)carve(ND4);
  float* hbuf = (float*)carve(ND4);
  float* qf   = (float*)carve(ND4);
  float* kf   = (float*)carve(ND4);
  float* vf   = (float*)carve(ND4);
  float* v1f  = (float*)carve(ND4);
  float* ybuf = (float*)carve(ND4);
  float* tbuf = (float*)carve((size_t)NTOK * FFDIM * 4);
  bf16_t* qbf  = (bf16_t*)carve(ND2);
  bf16_t* kbf  = (bf16_t*)carve(ND2);
  bf16_t* vtbf = (bf16_t*)carve(ND2);
  int*   levels = (int*)carve((size_t)NTOK * 4);
  float* sumexp = (float*)carve((size_t)NTOK * 4);
  float* tgtlog = (float*)carve((size_t)NTOK * 4);

  embed_kernel<<<NTOK, 256, 0, stream>>>(idx, wte, x, x0, levels);

  dim3 gD(DMODEL / 64, NTOK / 128);
  dim3 gFF(FFDIM / 64, NTOK / 128);
  for (int i = 0; i < NLAYER; ++i) {
    resid_rms_kernel<<<NTOK, 256, 0, stream>>>(x, x0, lambdas, i, xr, hbuf);

    const float* wqi = wq + (size_t)i * DMODEL * DMODEL;
    const float* wki = wk + (size_t)i * DMODEL * DMODEL;
    const float* wvi = wv + (size_t)i * DMODEL * DMODEL;
    const float* woi = wo + (size_t)i * DMODEL * DMODEL;
    const float* w1i = w1 + (size_t)i * FFDIM * DMODEL;
    const float* w2i = w2 + (size_t)i * DMODEL * FFDIM;

    gemm_kernel<0><<<gD, 256, 0, stream>>>(hbuf, wqi, nullptr, qf,
                                           NTOK, DMODEL, DMODEL);
    gemm_kernel<0><<<gD, 256, 0, stream>>>(hbuf, wki, nullptr, kf,
                                           NTOK, DMODEL, DMODEL);
    gemm_kernel<0><<<gD, 256, 0, stream>>>(hbuf, wvi, nullptr, vf,
                                           NTOK, DMODEL, DMODEL);

    qk_prep_kernel<<<dim3(NTOK, NHEAD), 32, 0, stream>>>(
        qf, kf, vf, v1f, lamb, i, qbf, kbf, vtbf);

    attn_kernel<<<dim3(SQ / 16, NHEAD, BATCH), 32, 0, stream>>>(
        qbf, kbf, vtbf, levels, ybuf);

    gemm_kernel<1><<<gD, 256, 0, stream>>>(ybuf, woi, xr, x,
                                           NTOK, DMODEL, DMODEL);

    rms_kernel<<<NTOK, 256, 0, stream>>>(x, hbuf);
    gemm_kernel<2><<<gFF, 256, 0, stream>>>(hbuf, w1i, nullptr, tbuf,
                                            NTOK, FFDIM, DMODEL);
    gemm_kernel<1><<<gD, 256, 0, stream>>>(tbuf, w2i, x, x,
                                           NTOK, DMODEL, FFDIM);
  }

  rms_kernel<<<NTOK, 256, 0, stream>>>(x, hbuf);
  hipMemsetAsync(sumexp, 0, (size_t)NTOK * 4, stream);
  lm_kernel<<<dim3((VOCAB + 63) / 64, NTOK / 128), 256, 0, stream>>>(
      hbuf, lm_head, target, sumexp, tgtlog, NTOK, VOCAB, DMODEL);
  loss_kernel<<<(NTOK + 255) / 256, 256, 0, stream>>>(
      target, sumexp, tgtlog, (float*)d_out, NTOK);
}
